// RIConv2SetAbstraction_68582037783101
// MI455X (gfx1250) — matrix operand order repaired and checked by reading, not compiled, not run on hardware
//
#include <hip/hip_runtime.h>

typedef _Float16 v16h __attribute__((ext_vector_type(16)));
typedef float    v8f  __attribute__((ext_vector_type(8)));

#define B_   4
#define N_   4096
#define K_   32
#define CP_  64
#define EPS_BN 1e-5f

// ---------------- workspace layout (bytes) ----------------
// ri (f16 packed, 4 u32 per row): B*N*K rows
#define RI_U32    (B_*N_*K_*4)                       // 2,097,152 u32
#define RI_OFF    0ull
#define IDXO_OFF  ((size_t)RI_U32*4)                 // int[B*N*K]
#define WF_OFF    (IDXO_OFF + (size_t)B_*N_*K_*4)    // weight fragments (u32), 16B aligned
// frag indices (each frag = 256 u32 = 32 lanes * 8 u32 = 64 uint4)
#define FR_P0 0
#define FR_P1 2
#define FR_M0 6
#define FR_M1 38
#define FR_TOT 102
#define BN_OFF (WF_OFF + (size_t)FR_TOT*256*4)
#define BN_S_P0 0
#define BN_S_P1 32
#define BN_S_M0 96
#define BN_S_M1 224
#define BN_TOT 480

// ---------------- prep: weight swizzle + BN fold ----------------
__global__ void prep_w(const float* __restrict__ W, int Cin, int nks,
                       unsigned int* __restrict__ out, int nfr) {
  int nt = blockDim.x * gridDim.x;
  for (int i = blockIdx.x * blockDim.x + threadIdx.x; i < nfr * 256; i += nt) {
    int f = i >> 8, r = i & 255, lane = r >> 3, v = r & 7;
    int ct = f / nks, ks = f - ct * nks;
    int h = lane >> 4, col = ct * 16 + (lane & 15);
    int k = ks * 32 + h * 16 + v * 2;
    float w0 = (k     < Cin) ? W[col * Cin + k]     : 0.f;
    float w1 = (k + 1 < Cin) ? W[col * Cin + k + 1] : 0.f;
    union { _Float16 h2[2]; unsigned int u; } cv;
    cv.h2[0] = (_Float16)w0; cv.h2[1] = (_Float16)w1;
    out[i] = cv.u;
  }
}

__global__ void prep_bn(const float* __restrict__ b, const float* __restrict__ g,
                        const float* __restrict__ be, const float* __restrict__ m,
                        const float* __restrict__ v, int C,
                        float* __restrict__ S, float* __restrict__ Bo) {
  int i = blockIdx.x * blockDim.x + threadIdx.x;
  if (i < C) {
    float s = g[i] / sqrtf(v[i] + EPS_BN);
    S[i]  = s;
    Bo[i] = (b[i] - m[i]) * s + be[i];
  }
}

// ---------------- KNN ----------------
#define KNN_T  128
#define KNN_CH 1024
__global__ __launch_bounds__(KNN_T) void knn_kernel(const float* __restrict__ xyz,
                                                    int* __restrict__ idx_out) {
  __shared__ float4 cand[KNN_CH];            // x,y,z,|p|^2  -> ds_load_b128
  __shared__ float hd[K_][KNN_T];
  __shared__ int   hi[K_][KNN_T];
  int tid = threadIdx.x;
  int b   = blockIdx.x / (N_ / KNN_T);
  int n   = (blockIdx.x % (N_ / KNN_T)) * KNN_T + tid;
  const float* xb = xyz + (size_t)b * N_ * 3;
  float qx = xb[n * 3], qy = xb[n * 3 + 1], qz = xb[n * 3 + 2];
  float q2 = qx * qx + qy * qy + qz * qz;
  for (int s = 0; s < K_; s++) { hd[s][tid] = 3.4e38f; hi[s][tid] = 0; }
  float wmax = 3.4e38f; int wpos = 0;
  for (int base = 0; base < N_; base += KNN_CH) {
    __syncthreads();
    for (int t = tid; t < KNN_CH; t += KNN_T) {
      float x = xb[(base + t) * 3], y = xb[(base + t) * 3 + 1], z = xb[(base + t) * 3 + 2];
      cand[t] = make_float4(x, y, z, x * x + y * y + z * z);
    }
    __syncthreads();
    for (int m = 0; m < KNN_CH; m++) {
      float4 c = cand[m];
      float d = q2 + c.w - 2.f * (qx * c.x + qy * c.y + qz * c.z);
      if (d < wmax) {
        hd[wpos][tid] = d; hi[wpos][tid] = base + m;
        float mx = -3.4e38f; int mp = 0;
        for (int s = 0; s < K_; s++) { float vv = hd[s][tid]; if (vv > mx) { mx = vv; mp = s; } }
        wmax = mx; wpos = mp;
      }
    }
  }
  // sort ascending by (dist, idx) to match top_k(-dist) ordering
  for (int i = 1; i < K_; i++) {
    float dv = hd[i][tid]; int iv = hi[i][tid];
    int j = i - 1;
    while (j >= 0 && (hd[j][tid] > dv || (hd[j][tid] == dv && hi[j][tid] > iv))) {
      hd[j + 1][tid] = hd[j][tid]; hi[j + 1][tid] = hi[j][tid]; j--;
    }
    hd[j + 1][tid] = dv; hi[j + 1][tid] = iv;
  }
  int* op = idx_out + ((size_t)b * N_ + n) * K_;
  for (int s = 0; s < K_; s++) op[s] = hi[s][tid];
}

// ---------------- rotation-invariant features (one wave32 per point) -------
__global__ __launch_bounds__(32) void feat_kernel(const float* __restrict__ xyz,
                                                  const float* __restrict__ norm,
                                                  const int* __restrict__ idx_in,
                                                  unsigned int* __restrict__ ri_out,
                                                  int* __restrict__ idxo_out) {
  int lane = threadIdx.x;
  int pt = blockIdx.x;
  int b = pt >> 12;
  int n = pt & (N_ - 1);
  const float* xb = xyz  + (size_t)b * N_ * 3;
  const float* nb = norm + (size_t)b * N_ * 3;
  float qx = xb[n * 3], qy = xb[n * 3 + 1], qz = xb[n * 3 + 2];
  float nx = nb[n * 3], ny = nb[n * 3 + 1], nz = nb[n * 3 + 2];
  int id = idx_in[(size_t)pt * K_ + lane];
  float lxx = xb[id * 3] - qx, lxy = xb[id * 3 + 1] - qy, lxz = xb[id * 3 + 2] - qz;
  float dp = lxx * nx + lxy * ny + lxz * nz;
  float px = lxx - dp * nx, py = lxy - dp * ny, pz = lxz - dp * nz;
  float pl = sqrtf(px * px + py * py + pz * pz);
  float inv = pl > 0.f ? 1.f / pl : 0.f;
  float ux = px * inv, uy = py * inv, uz = pz * inv;
  // argmax of projected length (first occurrence)
  int mi = 0; float best = -1.f;
  for (int j = 0; j < K_; j++) { float v = __shfl(pl, j, 32); if (v > best) { best = v; mi = j; } }
  float vx = __shfl(ux, mi, 32), vy = __shfl(uy, mi, 32), vz = __shfl(uz, mi, 32);
  float dots = ux * vx + uy * vy + uz * vz;
  float cxv = uy * vz - uz * vy, cyv = uz * vx - ux * vz, czv = ux * vy - uy * vx;
  float sdot = cxv * nx + cyv * ny + czv * nz;
  float sg = (sdot > 0.f) ? 1.f : ((sdot < 0.f) ? -1.f : 0.f);
  if (lane == 0) sg = 1.f;
  dots = sg * dots - (1.f - sg);
  // stable descending rank (argsort(-dots))
  int rank = 0;
  for (int j = 0; j < K_; j++) {
    float dj = __shfl(dots, j, 32);
    if (dj > dots || (dj == dots && j < lane)) rank++;
  }
  __shared__ float sdots[K_];
  __shared__ int   sidx[K_];
  sdots[rank] = dots; sidx[rank] = id;
  __syncthreads();
  int io = sidx[lane];
  float ds    = sdots[lane];
  float dsp   = sdots[(lane + K_ - 1) & (K_ - 1)];
  float dlast = sdots[K_ - 1];
  float ax = xb[io * 3] - qx, ay = xb[io * 3 + 1] - qy, az = xb[io * 3 + 2] - qz;
  float len = sqrtf(ax * ax + ay * ay + az * az);
  float il = len > 0.f ? 1.f / len : 0.f;
  float uxx = ax * il, uyy = ay * il, uzz = az * il;
  float gnx = nb[io * 3], gny = nb[io * 3 + 1], gnz = nb[io * 3 + 2];
  float a0 = uxx * nx + uyy * ny + uzz * nz;
  float a1 = uxx * gnx + uyy * gny + uzz * gnz;
  const float eps = 1e-7f;
  float cn = fminf(fmaxf(gnx * nx + gny * ny + gnz * nz, -1.f + eps), 1.f - eps);
  float an = ((a0 < a1) ? 1.f : -1.f) * acosf(cn);
  int pprev = (lane + K_ - 1) & (K_ - 1);
  float pax = __shfl(ax, pprev, 32), pay = __shfl(ay, pprev, 32), paz = __shfl(az, pprev, 32);
  float pgx = __shfl(gnx, pprev, 32), pgy = __shfl(gny, pprev, 32), pgz = __shfl(gnz, pprev, 32);
  float ivx = ax - pax, ivy = ay - pay, ivz = az - paz;
  float ivl = sqrtf(ivx * ivx + ivy * ivy + ivz * ivz);
  float iil = ivl > 0.f ? 1.f / ivl : 0.f;
  float iux = ivx * iil, iuy = ivy * iil, iuz = ivz * iil;
  float ia0 = iux * gnx + iuy * gny + iuz * gnz;
  float ia1 = iux * pgx + iuy * pgy + iuz * pgz;
  float cc = fminf(fmaxf(gnx * pgx + gny * pgy + gnz * pgz, -1.f + eps), 1.f - eps);
  float ia2 = ((ia0 < ia1) ? 1.f : -1.f) * acosf(cc);
  float pif = (lane == 0) ? (-3.f - dlast) : (ds - dsp);
  float f[8] = {len, pif, a0, a1, an, ia0, ia1, ia2};
  unsigned int* rp = ri_out + ((size_t)pt * K_ + lane) * 4;
  #pragma unroll
  for (int t = 0; t < 4; t++) {
    union { _Float16 h2[2]; unsigned int u; } cv;
    cv.h2[0] = (_Float16)f[2 * t]; cv.h2[1] = (_Float16)f[2 * t + 1];
    rp[t] = cv.u;
  }
  idxo_out[(size_t)pt * K_ + lane] = io;
}

// ---------------- fused MLP + mean (one wave32 per point) ----------------
__device__ inline v16h load_bfrag(const uint4* __restrict__ wf4, int frag, int lane) {
  union { uint4 q[2]; v16h h; } cv;
  const uint4* p = wf4 + frag * 64 + lane * 2;
  cv.q[0] = p[0]; cv.q[1] = p[1];
  return cv.h;
}

// A-fragment from row-major LDS activation [16 rows][128 halfs]:
// lane needs two contiguous 16-byte runs of its row.
__device__ inline v16h afrag_lds(const _Float16* act, int lane, int ks) {
  int h_ = lane >> 4, m = lane & 15;
  const _Float16* base = act + m * 128 + ks * 32 + h_ * 16;
  union { uint4 q[2]; v16h h; } cv;
  cv.q[0] = *(const uint4*)(base);
  cv.q[1] = *(const uint4*)(base + 8);
  return cv.h;
}

__global__ __launch_bounds__(32) void mlp_kernel(const float* __restrict__ points,
                                                 const uint4* __restrict__ ri4,
                                                 const int* __restrict__ idxo,
                                                 const uint4* __restrict__ wf4,
                                                 const float* __restrict__ bnS,
                                                 const float* __restrict__ bnB,
                                                 float* __restrict__ out) {
  __shared__ _Float16 actA[16 * 128] __attribute__((aligned(16)));
  __shared__ _Float16 actB[16 * 128] __attribute__((aligned(16)));
  int lane = threadIdx.x;
  int pt = blockIdx.x;
  int b = pt >> 12;
  int h_ = lane >> 4, nn_ = lane & 15;
  float colsum[16];
  #pragma unroll
  for (int i = 0; i < 16; i++) colsum[i] = 0.f;

  for (int half_t = 0; half_t < 2; half_t++) {
    int rowBase = pt * K_ + half_t * 16;
    __syncthreads();
    // ---- p0: 8 -> 32 ----
    v16h a0;
    {
      union { uint4 q[2]; v16h h; } cv;
      cv.q[0] = make_uint4(0u, 0u, 0u, 0u);
      cv.q[1] = make_uint4(0u, 0u, 0u, 0u);
      if (lane < 16) cv.q[0] = ri4[rowBase + nn_];   // 8 features, one b128 load
      a0 = cv.h;
    }
    #pragma unroll
    for (int ct = 0; ct < 2; ct++) {
      v16h bf = load_bfrag(wf4, FR_P0 + ct, lane);
      v8f c = {0.f, 0.f, 0.f, 0.f, 0.f, 0.f, 0.f, 0.f};
      v8f d = __builtin_amdgcn_wmma_f32_16x16x32_f16(false, a0, false, bf, (short)0, c, false, false);
      int ch = ct * 16 + nn_;
      float s = bnS[BN_S_P0 + ch], bi = bnB[BN_S_P0 + ch];
      #pragma unroll
      for (int r = 0; r < 8; r++)
        actA[(h_ * 8 + r) * 128 + ch] = (_Float16)fmaxf(d[r] * s + bi, 0.f);
    }
    __syncthreads();
    // ---- p1: 32 -> 64 ---- (A loaded once, reused across 4 col-tiles)
    {
      v16h a = afrag_lds(actA, lane, 0);
      #pragma unroll
      for (int ct = 0; ct < 4; ct++) {
        v16h bf = load_bfrag(wf4, FR_P1 + ct, lane);
        v8f c = {0.f, 0.f, 0.f, 0.f, 0.f, 0.f, 0.f, 0.f};
        v8f d = __builtin_amdgcn_wmma_f32_16x16x32_f16(false, a, false, bf, (short)0, c, false, false);
        int ch = ct * 16 + nn_;
        float s = bnS[BN_S_P1 + ch], bi = bnB[BN_S_P1 + ch];
        #pragma unroll
        for (int r = 0; r < 8; r++)
          actB[(h_ * 8 + r) * 128 + ch] = (_Float16)fmaxf(d[r] * s + bi, 0.f);
      }
    }
    // gather neighbor point features into actB cols 64..127
    // float4 global load -> 4 packed halves -> one 8-byte LDS store
    {
      const int* iop = idxo + rowBase;
      for (int t = lane; t < 16 * (CP_ / 4); t += 32) {
        int row = t >> 4, cq = t & 15;
        int io = iop[row];
        float4 f4 = ((const float4*)(points + ((size_t)b * N_ + io) * CP_))[cq];
        union { _Float16 h4[4]; unsigned long long u; } cv;
        cv.h4[0] = (_Float16)f4.x; cv.h4[1] = (_Float16)f4.y;
        cv.h4[2] = (_Float16)f4.z; cv.h4[3] = (_Float16)f4.w;
        *(unsigned long long*)(actB + row * 128 + 64 + cq * 4) = cv.u;
      }
    }
    __syncthreads();
    // ---- m0: 128 -> 128 ---- (hoist 4 A-frags, reuse over 8 col-tiles)
    {
      v16h aB[4];
      #pragma unroll
      for (int ks = 0; ks < 4; ks++) aB[ks] = afrag_lds(actB, lane, ks);
      for (int ct = 0; ct < 8; ct++) {
        v8f acc = {0.f, 0.f, 0.f, 0.f, 0.f, 0.f, 0.f, 0.f};
        #pragma unroll
        for (int ks = 0; ks < 4; ks++) {
          v16h bf = load_bfrag(wf4, FR_M0 + ct * 4 + ks, lane);
          acc = __builtin_amdgcn_wmma_f32_16x16x32_f16(false, aB[ks], false, bf, (short)0, acc, false, false);
        }
        int ch = ct * 16 + nn_;
        float s = bnS[BN_S_M0 + ch], bi = bnB[BN_S_M0 + ch];
        #pragma unroll
        for (int r = 0; r < 8; r++)
          actA[(h_ * 8 + r) * 128 + ch] = (_Float16)fmaxf(acc[r] * s + bi, 0.f);
      }
    }
    __syncthreads();
    // ---- m1: 128 -> 256, BN+ReLU, reduce over rows (mean over K) ----
    {
      v16h aA[4];
      #pragma unroll
      for (int ks = 0; ks < 4; ks++) aA[ks] = afrag_lds(actA, lane, ks);
      for (int ct = 0; ct < 16; ct++) {
        v8f acc = {0.f, 0.f, 0.f, 0.f, 0.f, 0.f, 0.f, 0.f};
        #pragma unroll
        for (int ks = 0; ks < 4; ks++) {
          v16h bf = load_bfrag(wf4, FR_M1 + ct * 4 + ks, lane);
          acc = __builtin_amdgcn_wmma_f32_16x16x32_f16(false, aA[ks], false, bf, (short)0, acc, false, false);
        }
        int ch = ct * 16 + nn_;
        float s = bnS[BN_S_M1 + ch], bi = bnB[BN_S_M1 + ch];
        float part = 0.f;
        #pragma unroll
        for (int r = 0; r < 8; r++) part += fmaxf(acc[r] * s + bi, 0.f);
        part += __shfl_xor(part, 16, 32);
        colsum[ct] += part;
      }
    }
  }
  if (lane < 16) {
    float* op = out + (size_t)pt * 256;
    #pragma unroll
    for (int ct = 0; ct < 16; ct++) op[ct * 16 + lane] = colsum[ct] * (1.f / 32.f);
  }
}

// ---------------- launcher ----------------
extern "C" void kernel_launch(void* const* d_in, const int* in_sizes, int n_in,
                              void* d_out, int out_size, void* d_ws, size_t ws_size,
                              hipStream_t stream) {
  (void)in_sizes; (void)n_in; (void)out_size; (void)ws_size;
  const float* xyz    = (const float*)d_in[0];
  const float* norm   = (const float*)d_in[1];
  const float* points = (const float*)d_in[2];
  const float* W[4] = {(const float*)d_in[3], (const float*)d_in[9],
                       (const float*)d_in[15], (const float*)d_in[21]};
  float* out = (float*)d_out;
  int* idx_out = (int*)((float*)d_out + (size_t)B_ * N_ * 256);

  char* ws = (char*)d_ws;
  unsigned int* ri   = (unsigned int*)(ws + RI_OFF);
  int*          idxo = (int*)(ws + IDXO_OFF);
  unsigned int* wf   = (unsigned int*)(ws + WF_OFF);
  float*        bnS  = (float*)(ws + BN_OFF);
  float*        bnB  = bnS + BN_TOT;

  const int nks[4]    = {1, 1, 4, 4};
  const int cout[4]   = {32, 64, 128, 256};
  const int cin[4]    = {8, 32, 128, 128};
  const int frBase[4] = {FR_P0, FR_P1, FR_M0, FR_M1};
  const int bnBase[4] = {BN_S_P0, BN_S_P1, BN_S_M0, BN_S_M1};
  for (int L = 0; L < 4; L++) {
    int nfr = (cout[L] / 16) * nks[L];
    prep_w<<<dim3((nfr * 256 + 255) / 256), 256, 0, stream>>>(
        W[L], cin[L], nks[L], wf + frBase[L] * 256, nfr);
    prep_bn<<<dim3((cout[L] + 63) / 64), 64, 0, stream>>>(
        (const float*)d_in[3 + 6 * L + 1], (const float*)d_in[3 + 6 * L + 2],
        (const float*)d_in[3 + 6 * L + 3], (const float*)d_in[3 + 6 * L + 4],
        (const float*)d_in[3 + 6 * L + 5], cout[L], bnS + bnBase[L], bnB + bnBase[L]);
  }
  knn_kernel<<<dim3(B_ * (N_ / KNN_T)), KNN_T, 0, stream>>>(xyz, idx_out);
  feat_kernel<<<dim3(B_ * N_), 32, 0, stream>>>(xyz, norm, idx_out, ri, idxo);
  mlp_kernel<<<dim3(B_ * N_), 32, 0, stream>>>(points, (const uint4*)ri, idxo,
                                               (const uint4*)wf, bnS, bnB, out);
}
